// E2EBoltConventionalTraining_86706799772094
// MI455X (gfx1250) — matrix operand order repaired and checked by reading, not compiled, not run on hardware
//
#include <hip/hip_runtime.h>
#include <cstdint>

typedef __attribute__((ext_vector_type(16))) __bf16 v16bf;
typedef __attribute__((ext_vector_type(2)))  __bf16 bf16x2;
typedef __attribute__((ext_vector_type(8)))  float  v8f;
typedef __attribute__((ext_vector_type(2)))  float  v2f;

#define NWAVES 8
#define TPW    8          // tiles (of 16 rows) per wave
#define HID    128
#define HP     136        // padded row stride (ushorts) to break LDS bank conflicts
#define NB     6

union Frag { v16bf v; uint4 q[2]; };

__device__ __forceinline__ unsigned short f2bf(float f) {
    union { __bf16 h; unsigned short u; } cv;
    cv.h = (__bf16)f;                      // native cvt, RNE
    return cv.u;
}

__device__ __forceinline__ unsigned int pk_bf16(float a, float b) {
    union { bf16x2 h; unsigned int u; } cv;
    v2f f; f.x = a; f.y = b;
    cv.h = __builtin_convertvector(f, bf16x2);   // v_cvt_pk_bf16_f32
    return cv.u;
}

// ReLU + f32->bf16 + column store of one 16x16 C-fragment into [m][k] LDS tile
__device__ __forceinline__ void store_col_relu(unsigned short* dst, int col, int hi8,
                                               const v8f& c, float bias) {
    #pragma unroll
    for (int r = 0; r < 8; r += 2) {
        float v0 = fmaxf(c[r]     + bias, 0.f);
        float v1 = fmaxf(c[r + 1] + bias, 0.f);
        unsigned int pk = pk_bf16(v0, v1);
        dst[(r     + hi8) * HP + col] = (unsigned short)pk;
        dst[(r + 1 + hi8) * HP + col] = (unsigned short)(pk >> 16);  // b16_d16_hi
    }
}

__global__ __launch_bounds__(256) void demapper_mlp_fused(
    const float* __restrict__ z,  const float* __restrict__ W1, const float* __restrict__ b1,
    const float* __restrict__ W2, const float* __restrict__ b2,
    const float* __restrict__ W3, const float* __restrict__ b3,
    float* __restrict__ out)
{
    __shared__ unsigned short sW2T[HID * HP];        // bf16, [n][k] (transposed for B-frag reads)
    __shared__ unsigned short sW3T[16 * HP];         // bf16, [o][k], outputs padded 6 -> 16
    __shared__ unsigned short sH[NWAVES][16 * HP];   // per-wave activation scratch, bf16 [m][k]
    __shared__ float sB2[HID];
    __shared__ float sB3[8];

    const int tid  = threadIdx.x;
    const int wave = tid >> 5;
    const int lane = tid & 31;
    const int l16  = lane & 15;
    const bool hiL = lane >= 16;
    const int hi8  = hiL ? 8 : 0;     // A-frag / C-frag row split
    const int hi16 = hiL ? 16 : 0;    // bf16 B-frag K split

    // ---------------- one-time LDS staging (f32 -> bf16) ----------------
    for (int i = tid; i < HID * HID; i += 256) {
        int h = i >> 7, n = i & (HID - 1);
        sW2T[n * HP + h] = f2bf(W2[i]);                 // W2 is [h][n] row-major
    }
    for (int i = tid; i < 16 * HP; i += 256) sW3T[i] = 0;
    for (int i = tid; i < HID;     i += 256) sB2[i] = b2[i];
    if (tid < NB) sB3[tid] = b3[tid];
    __syncthreads();                                    // zero-fill done before W3 scatter
    for (int i = tid; i < HID * NB; i += 256) {
        int h = i / NB, o = i - h * NB;
        sW3T[o * HP + h] = f2bf(W3[i]);                 // W3 is [h][o] row-major
    }
    __syncthreads();

    unsigned short* const sHw = &sH[wave][0];

    // ---------------- hoisted per-lane constants ----------------
    // Layer-1 B fragments (f32 4x16 per N-tile): rows = {W1[0], W1[1], W1[2], b1}.
    // f32 B layout: lanes 0-15 hold K=0,1 ; lanes 16-31 hold K=2,3 ; N = lane%16.
    v2f w1b[8];
    #pragma unroll
    for (int nt = 0; nt < 8; ++nt) {
        const int n = nt * 16 + l16;
        w1b[nt].x = hiL ? W1[2 * HID + n] : W1[0 * HID + n];
        w1b[nt].y = hiL ? b1[n]           : W1[1 * HID + n];
    }

    // Layer-3 B fragments (bf16): held in regs for the whole kernel
    Frag b3f[4];
    #pragma unroll
    for (int kf = 0; kf < 4; ++kf) {
        const uint4* p = (const uint4*)&sW3T[l16 * HP + kf * 32 + hi16];
        b3f[kf].q[0] = p[0];
        b3f[kf].q[1] = p[1];
    }

    // Layer-2 B fragments for N-tiles 0..3: loop-invariant -> hold in registers
    // (half hoisted: +128 VGPRs, keeps occupancy while halving LDS frag traffic)
    Frag b2f[16];
    #pragma unroll
    for (int nt = 0; nt < 4; ++nt) {
        #pragma unroll
        for (int kf = 0; kf < 4; ++kf) {
            const uint4* p = (const uint4*)&sW2T[(nt * 16 + l16) * HP + kf * 32 + hi16];
            b2f[nt * 4 + kf].q[0] = p[0];
            b2f[nt * 4 + kf].q[1] = p[1];
        }
    }

    const int waveRow0 = (blockIdx.x * (NWAVES * TPW) + wave * TPW) * 16;

    // Software-pipelined z load: issue tile t+1's load during tile t's compute
    float zp0, zp1, zp2;
    {
        const float* zp = z + (size_t)(waveRow0 + l16) * 3;
        zp0 = zp[0]; zp1 = zp[1]; zp2 = zp[2];
    }

    // ---------------- main loop: 8 tiles of 16 rows per wave ----------------
    for (int t = 0; t < TPW; ++t) {
        const int rowBase = waveRow0 + t * 16;

        // consume previous prefetch (branchless A-fragment select)
        v2f az;
        az.x = hiL ? zp2  : zp0;
        az.y = hiL ? 1.0f : zp1;

        // issue next tile's z load now; its s_wait lands a full tile later
        if (t + 1 < TPW) {
            const float* zp = z + (size_t)(rowBase + 16 + l16) * 3;
            zp0 = zp[0]; zp1 = zp[1]; zp2 = zp[2];
        }

        // ---- layer 1 via V_WMMA_F32_16X16X4_F32 ----
        // A (16x4 f32): row m = [z0, z1, z2, 1.0]; documented layout:
        // lanes 0-15: VGPR0=K0, VGPR1=K1 ; lanes 16-31: VGPR0=K2, VGPR1=K3.
        #pragma unroll
        for (int nt = 0; nt < 8; ++nt) {
            v8f c1 = {0.f, 0.f, 0.f, 0.f, 0.f, 0.f, 0.f, 0.f};
            c1 = __builtin_amdgcn_wmma_f32_16x16x4_f32(
                     false, az, false, w1b[nt], (short)0, c1, false, false);
            store_col_relu(sHw, nt * 16 + l16, hi8, c1, 0.f);   // bias already folded
        }

        // ---- load A fragments (16x32 bf16 layout) for layer 2 ----
        Frag afr[4];
        #pragma unroll
        for (int kf = 0; kf < 4; ++kf) {
            const uint4* p = (const uint4*)&sHw[l16 * HP + kf * 32 + hi16];
            afr[kf].q[0] = p[0];      // K = k0 .. k0+7
            afr[kf].q[1] = p[1];      // K = k0+8 .. k0+15
        }

        // ---- layer 2: 8 N-tiles x 4 K-frags of v_wmma_f32_16x16x32_bf16 ----
        #pragma unroll
        for (int nt = 0; nt < 8; ++nt) {
            v8f c = {0.f, 0.f, 0.f, 0.f, 0.f, 0.f, 0.f, 0.f};
            #pragma unroll
            for (int kf = 0; kf < 4; ++kf) {
                if (nt < 4) {
                    c = __builtin_amdgcn_wmma_f32_16x16x32_bf16(
                            false, afr[kf].v, false, b2f[nt * 4 + kf].v,
                            (short)0, c, false, false);
                } else {
                    const uint4* p = (const uint4*)&sW2T[(nt * 16 + l16) * HP + kf * 32 + hi16];
                    Frag bf; bf.q[0] = p[0]; bf.q[1] = p[1];   // K = kb..kb+15 sequential
                    c = __builtin_amdgcn_wmma_f32_16x16x32_bf16(
                            false, afr[kf].v, false, bf.v, (short)0, c, false, false);
                }
            }
            store_col_relu(sHw, nt * 16 + l16, hi8, c, sB2[nt * 16 + l16]);
        }

        // ---- layer 3: h2 (A) x W3^T (B, hoisted) ----
        Frag a2[4];
        #pragma unroll
        for (int kf = 0; kf < 4; ++kf) {
            const uint4* p = (const uint4*)&sHw[l16 * HP + kf * 32 + hi16];
            a2[kf].q[0] = p[0];
            a2[kf].q[1] = p[1];
        }
        v8f c3 = {0.f, 0.f, 0.f, 0.f, 0.f, 0.f, 0.f, 0.f};
        #pragma unroll
        for (int kf = 0; kf < 4; ++kf) {
            c3 = __builtin_amdgcn_wmma_f32_16x16x32_bf16(
                    false, a2[kf].v, false, b3f[kf].v, (short)0, c3, false, false);
        }

        // ---- epilogue: +b3, write only the 6 real output columns ----
        if (l16 < NB) {
            const float bo = sB3[l16];
            #pragma unroll
            for (int r = 0; r < 8; ++r) {
                const int m = r + hi8;
                out[(size_t)(rowBase + m) * NB + l16] = c3[r] + bo;
            }
        }
    }
}

extern "C" void kernel_launch(void* const* d_in, const int* in_sizes, int n_in,
                              void* d_out, int out_size, void* d_ws, size_t ws_size,
                              hipStream_t stream) {
    const float* z  = (const float*)d_in[0];
    const float* W1 = (const float*)d_in[1];
    const float* b1 = (const float*)d_in[2];
    const float* W2 = (const float*)d_in[3];
    const float* b2 = (const float*)d_in[4];
    const float* W3 = (const float*)d_in[5];
    const float* b3 = (const float*)d_in[6];
    float* out = (float*)d_out;

    // 2,097,152 rows / (8 waves * 8 tiles * 16 rows) = 2048 workgroups
    dim3 grid(2048), block(256);
    hipLaunchKernelGGL(demapper_mlp_fused, grid, block, 0, stream,
                       z, W1, b1, W2, b2, W3, b3, out);
}
